// MultiHeadAttention_4432406249987
// MI455X (gfx1250) — matrix operand order repaired and checked by reading, not compiled, not run on hardware
//
#include <hip/hip_runtime.h>
#include <hip/hip_bf16.h>

// ---------------------------------------------------------------------------
// Multi-head attention forward for MI455X (gfx1250).
// bf16 WMMA matmuls, fp32 accumulation; softmax / layernorm in fp32.
// GEMM and attn@V stage shared operand slabs in LDS via the CDNA5
// async-to-LDS DMA engine with double buffering (ASYNCcnt pipelining).
// ---------------------------------------------------------------------------

typedef __attribute__((ext_vector_type(16))) __bf16 v16bf;
typedef __attribute__((ext_vector_type(8)))  float  v8f;

#define NUM_HEADS  16
#define EMBED_DIM  1024
#define HEAD_DIM   64
#define BATCH      4
#define SEQ        1024
#define NROWS      (BATCH * SEQ)      // 4096
#define LN_EPS     1e-5f

#define GEMM_BM 128   // rows per block (8 waves x 16)
#define GEMM_BN 64    // cols per block (4 WMMA tiles per wave)
#define GEMM_BK 32    // k-step (one WMMA K)
#define GEMM_SLAB (GEMM_BK * GEMM_BN) // 2048 fp32 = 8KB

__device__ __forceinline__ __bf16 to_bf16(float x)  { return (__bf16)x; }
__device__ __forceinline__ __bf16 to_bf16(__bf16 x) { return x; }

// ---------------------------------------------------------------------------
// Fragment builders for CDNA5 wave32 WMMA; A/B use contiguous k = 16h + s, C/D per
// (CDNA5 ISA §7.12.2).
// ---------------------------------------------------------------------------

// A-matrix 16x32 (MxK), 16-bit. Lane L<16: M=L, VGPR j holds K={2j,2j+1};
// lanes 16-31: M=L-16, K base +16.
template <typename T>
__device__ __forceinline__ v16bf load_frag_a(const T* __restrict__ A, int lda,
                                             int row0, int k0) {
    const int lane = threadIdx.x & 31;
    const int m    = lane & 15;
    const int kb   = (lane >> 4) ? 16 : 0;
    const T* p = A + (long)(row0 + m) * lda + k0;
    v16bf a;
#pragma unroll
    for (int j = 0; j < 8; ++j) {
        const int k = kb + 2 * j;
        a[2 * j + 0] = to_bf16(p[k + 0]);
        a[2 * j + 1] = to_bf16(p[k + 1]);
    }
    return a;
}

// B-matrix 32x16 (KxN), 16-bit. Lane L<16: N=L, VGPR j holds K={2j,2j+1};
// lanes 16-31: N=L-16, K base +16.
// TRANS=true : element B[k][n] = Src[(col0+n)*ldb + (k0+k)]  (B = Src^T).
template <typename T, bool TRANS>
__device__ __forceinline__ v16bf load_frag_b(const T* __restrict__ Bsrc, int ldb,
                                             int k0, int col0) {
    const int lane = threadIdx.x & 31;
    const int n    = lane & 15;
    const int kb   = (lane >> 4) ? 16 : 0;
    v16bf b;
#pragma unroll
    for (int j = 0; j < 8; ++j) {
        const int k = kb + 2 * j;
        if (TRANS) {
            const T* p = Bsrc + (long)(col0 + n) * ldb + (k0 + k);
            b[2 * j + 0] = to_bf16(p[0]);
            b[2 * j + 1] = to_bf16(p[1]);
        } else {
            b[2 * j + 0] = to_bf16(Bsrc[(long)(k0 + k + 0) * ldb + col0 + n]);
            b[2 * j + 1] = to_bf16(Bsrc[(long)(k0 + k + 1) * ldb + col0 + n]);
        }
    }
    return b;
}

// B fragment from an LDS-resident fp32 slab [GEMM_BK x GEMM_BN], row-major.
__device__ __forceinline__ v16bf lds_frag_b_f32(const float* __restrict__ wt,
                                                int colbase) {
    const int lane = threadIdx.x & 31;
    const int n    = lane & 15;
    const int kb   = (lane >> 4) ? 16 : 0;
    v16bf b;
#pragma unroll
    for (int j = 0; j < 8; ++j) {
        const int k = kb + 2 * j;
        b[2 * j + 0] = to_bf16(wt[(k + 0) * GEMM_BN + colbase + n]);
        b[2 * j + 1] = to_bf16(wt[(k + 1) * GEMM_BN + colbase + n]);
    }
    return b;
}

// B fragment from an LDS-resident bf16 slab [32 x 16], row-major.
__device__ __forceinline__ v16bf lds_frag_b_bf16(const __bf16* __restrict__ vt) {
    const int lane = threadIdx.x & 31;
    const int n    = lane & 15;
    const int kb   = (lane >> 4) ? 16 : 0;
    v16bf b;
#pragma unroll
    for (int j = 0; j < 8; ++j) {
        const int k = kb + 2 * j;
        b[2 * j + 0] = vt[(k + 0) * 16 + n];
        b[2 * j + 1] = vt[(k + 1) * 16 + n];
    }
    return b;
}

// C/D 16x16 fp32: lane L<16 -> N=L, VGPR r -> M=r; lanes 16-31 -> M=8+r.
template <typename TOUT>
__device__ __forceinline__ void store_frag_c(TOUT* __restrict__ C, int ldc,
                                             int row0, int col0, v8f c,
                                             float scale) {
    const int lane = threadIdx.x & 31;
    const int n    = lane & 15;
    const int mb   = (lane >> 4) << 3;
#pragma unroll
    for (int r = 0; r < 8; ++r)
        C[(long)(row0 + mb + r) * ldc + col0 + n] = (TOUT)(c[r] * scale);
}

__device__ __forceinline__ v8f wmma_bf16(v16bf a, v16bf b, v8f c) {
    return __builtin_amdgcn_wmma_f32_16x16x32_bf16(
        /*neg_a=*/false, a, /*neg_b=*/false, b,
        /*c_mod=*/(short)0, c, /*reuse_a=*/false, /*reuse_b=*/false);
}

// CDNA5 async DMA global -> LDS, tracked by ASYNCcnt (in-order completion).
// (CDNA5 ISA; inline asm — portable across toolchains.)
__device__ __forceinline__ void async_copy_b128(unsigned lds_byte_addr,
                                                const void* gaddr) {
    asm volatile("global_load_async_to_lds_b128 %0, %1, off"
                 :: "v"(lds_byte_addr), "v"(gaddr) : "memory");
}
__device__ __forceinline__ void async_copy_b32(unsigned lds_byte_addr,
                                               const void* gaddr) {
    asm volatile("global_load_async_to_lds_b32 %0, %1, off"
                 :: "v"(lds_byte_addr), "v"(gaddr) : "memory");
}
__device__ __forceinline__ void wait_asynccnt0() {
    asm volatile("s_wait_asynccnt 0" ::: "memory");
}
__device__ __forceinline__ void wait_asynccnt1() {
    asm volatile("s_wait_asynccnt 1" ::: "memory");
}
__device__ __forceinline__ void wait_asynccnt2() {
    asm volatile("s_wait_asynccnt 2" ::: "memory");
}

// ---------------------------------------------------------------------------
// Kernel 1/5: X[MxK] @ W[KxN] -> C[MxN].
// Block = 256 threads (8 waves) computes a 128x64 macro-tile: wave w owns the
// 16-row strip row0+16w and 4 accumulators across 64 columns (one A fragment
// feeds 4 WMMAs).  The W k-slab [32 x 64] fp32 (8KB) is DMA'd into a
// double-buffered LDS slab with global_load_async_to_lds_b128: while slab s
// is consumed, slab s+1 is in flight (s_wait_asynccnt 2 drains only the
// older pair — async completions are in-order per wave).
// ---------------------------------------------------------------------------
template <typename TA, typename TOUT>
__global__ __launch_bounds__(256) void gemm_xw(const TA* __restrict__ A,
                                               const float* __restrict__ W,
                                               TOUT* __restrict__ C,
                                               int M, int N, int K) {
    __shared__ float wtile[2][GEMM_SLAB];        // 2 x 8 KB

    const int tid  = threadIdx.x;
    const int wave = tid >> 5;
    const int tiles_n = N / GEMM_BN;
    const int bm = blockIdx.x / tiles_n;
    const int bn = blockIdx.x % tiles_n;
    const int row0 = bm * GEMM_BM + wave * 16;
    const int col0 = bn * GEMM_BN;

    // LDS byte address: flat-LDS aperture keeps the offset in addr[31:0].
    const unsigned lds_base  = (unsigned)(size_t)(const void*)&wtile[0][0];
    const unsigned slab_bytes = GEMM_SLAB * 4u;

    // Each thread DMAs 8 floats (2 x b128) of the 32x64 slab per k-step.
    const int li = tid * 8;          // linear element index in slab
    const int lr = li >> 6;          // slab row (0..31)
    const int lc = li & 63;          // slab col (0..56 step 8)

    v8f acc0 = {}, acc1 = {}, acc2 = {}, acc3 = {};

    const int nsteps = K / GEMM_BK;

    // Prologue: DMA slab 0 into buffer 0.
    {
        const float* gsrc = W + (long)lr * N + col0 + lc;
        const unsigned ldst = lds_base + (unsigned)li * 4u;
        async_copy_b128(ldst,       (const void*)gsrc);
        async_copy_b128(ldst + 16u, (const void*)(gsrc + 4));
    }

    for (int s = 0; s < nsteps; ++s) {
        const int k0  = s * GEMM_BK;
        const int cur = s & 1;

        if (s + 1 < nsteps) {
            // DMA the next slab into the alternate buffer, then drain only
            // the older pair: double-buffered overlap of DMA and WMMA.
            const float* gsrc = W + (long)(k0 + GEMM_BK + lr) * N + col0 + lc;
            const unsigned ldst = lds_base + (cur ^ 1) * slab_bytes
                                           + (unsigned)li * 4u;
            async_copy_b128(ldst,       (const void*)gsrc);
            async_copy_b128(ldst + 16u, (const void*)(gsrc + 4));
            wait_asynccnt2();
        } else {
            wait_asynccnt0();
        }

        if (k0 + GEMM_BK < K) {       // global_prefetch_b8 of next A slab
            const TA* pa = A + (long)(row0 + (tid & 15)) * K + k0 + GEMM_BK;
            __builtin_prefetch((const void*)pa, 0, 0);
        }

        v16bf a = load_frag_a<TA>(A, K, row0, k0);

        __syncthreads();              // all waves' slab s landed

        v16bf b0 = lds_frag_b_f32(wtile[cur], 0);
        v16bf b1 = lds_frag_b_f32(wtile[cur], 16);
        v16bf b2 = lds_frag_b_f32(wtile[cur], 32);
        v16bf b3 = lds_frag_b_f32(wtile[cur], 48);
        acc0 = wmma_bf16(a, b0, acc0);
        acc1 = wmma_bf16(a, b1, acc1);
        acc2 = wmma_bf16(a, b2, acc2);
        acc3 = wmma_bf16(a, b3, acc3);

        __syncthreads();              // slab s fully consumed before reuse
    }

    store_frag_c<TOUT>(C, N, row0, col0 +  0, acc0, 1.0f);
    store_frag_c<TOUT>(C, N, row0, col0 + 16, acc1, 1.0f);
    store_frag_c<TOUT>(C, N, row0, col0 + 32, acc2, 1.0f);
    store_frag_c<TOUT>(C, N, row0, col0 + 48, acc3, 1.0f);
}

// ---------------------------------------------------------------------------
// Kernel 2: scores = (Q K^T) / sqrt(64) per (head, batch), fp32 into the attn
// region of d_out.  K-dim = HEAD_DIM = 64 -> 2 WMMAs per 16x16 tile.
// attn layout: [H, B, S, S].  Both fragments read contiguous bf16 (the
// compiler vectorizes them to global_load_b128 clauses).
// ---------------------------------------------------------------------------
__global__ __launch_bounds__(256) void scores_wmma(const __bf16* __restrict__ Qb,
                                                   const __bf16* __restrict__ Kb,
                                                   float* __restrict__ attn) {
    const int  wave = threadIdx.x >> 5;
    const long tile = (long)blockIdx.x * 8 + wave;  // total = 64*64*64 = 262144
    const int  hb   = (int)(tile >> 12);            // 4096 tiles per (h,b)
    const int  rem  = (int)(tile & 4095);
    const int  tq   = rem >> 6;
    const int  tk   = rem & 63;
    const int  h    = hb >> 2;
    const int  b    = hb & 3;

    const __bf16* Qh = Qb + (long)b * SEQ * EMBED_DIM + h * HEAD_DIM;
    const __bf16* Kh = Kb + (long)b * SEQ * EMBED_DIM + h * HEAD_DIM;

    v8f acc = {};
#pragma unroll
    for (int k0 = 0; k0 < HEAD_DIM; k0 += 32) {
        v16bf a  = load_frag_a<__bf16>(Qh, EMBED_DIM, tq * 16, k0);
        v16bf bb = load_frag_b<__bf16, true>(Kh, EMBED_DIM, k0, tk * 16);
        acc = wmma_bf16(a, bb, acc);
    }
    float* Ch = attn + (long)hb * SEQ * SEQ;
    store_frag_c<float>(Ch, SEQ, tq * 16, tk * 16, acc, 0.125f); // 1/sqrt(64)
}

// ---------------------------------------------------------------------------
// Kernel 3: row softmax with post-hoc causal renormalization == softmax over
// keys k <= q with zeros elsewhere.  One 256-thread block per row.
// ---------------------------------------------------------------------------
__global__ __launch_bounds__(256) void softmax_causal(float* __restrict__ attn,
                                                      const int* __restrict__ maskp) {
    const long row = blockIdx.x;                 // hb*S + q, total 65536
    const int  q   = (int)(row & (SEQ - 1));
    float* p = attn + row * (long)SEQ;
    const int msk = *maskp;
    const int L   = msk ? (q + 1) : SEQ;
    const int t   = threadIdx.x;

    __shared__ float red[256];

    float m = -3.4e38f;
    for (int j = t; j < L; j += 256) m = fmaxf(m, p[j]);
    red[t] = m;
    __syncthreads();
    for (int s = 128; s > 0; s >>= 1) {
        if (t < s) red[t] = fmaxf(red[t], red[t + s]);
        __syncthreads();
    }
    m = red[0];
    __syncthreads();

    float sum = 0.0f;
    for (int j = t; j < L; j += 256) {
        const float e = __expf(p[j] - m);
        p[j] = e;
        sum += e;
    }
    red[t] = sum;
    __syncthreads();
    for (int s = 128; s > 0; s >>= 1) {
        if (t < s) red[t] += red[t + s];
        __syncthreads();
    }
    const float inv = 1.0f / red[0];

    for (int j = t; j < L; j += 256) p[j] *= inv;
    for (int j = L + t; j < SEQ; j += 256) p[j] = 0.0f;  // strict upper triangle
}

// ---------------------------------------------------------------------------
// Kernel 4: context = attn @ V per (head, batch).
// Block = 256 threads: one (h,b), 8 consecutive q-tiles (128 rows), one
// 16-column tile.  All 8 waves share the same V k-slab [32 x 16] bf16 (1KB),
// DMA'd via global_load_async_to_lds_b32 with the same double-buffered
// ASYNCcnt pipeline.  Causal: block k-loop stops at (qb+1)*128; each wave
// additionally skips k-steps beyond its own diagonal (uniform per wave, so
// EXEC stays all-ones for WMMA).  attn read fp32 -> bf16 in the A loader.
// ---------------------------------------------------------------------------
__global__ __launch_bounds__(256) void av_wmma(const float* __restrict__ attn,
                                               const __bf16* __restrict__ Vb,
                                               __bf16* __restrict__ concat,
                                               const int* __restrict__ maskp) {
    __shared__ __bf16 vtile[2][32 * 16];          // 2 x 1 KB

    const int tid  = threadIdx.x;
    const int wave = tid >> 5;
    const int bx   = blockIdx.x;                  // 2048 blocks
    const int hb   = bx >> 5;                     // (h,b): 64
    const int qb   = (bx >> 2) & 7;               // q block of 128 rows
    const int tn   = bx & 3;                      // 16-col tile in head
    const int h    = hb >> 2;
    const int b    = hb & 3;
    const int tq   = qb * 8 + wave;               // this wave's 16-row q tile

    const float*  Ah = attn + (long)hb * SEQ * SEQ;
    const __bf16* Vh = Vb + (long)b * SEQ * EMBED_DIM + h * HEAD_DIM + tn * 16;

    const int msk = *maskp;
    const int kend_blk  = msk ? (qb + 1) * 128 : SEQ;             // mult of 32
    const int kend_wave = msk ? ((tq * 16 + 16 + 31) & ~31) : SEQ;
    const int nsteps = kend_blk / 32;

    const unsigned lds_base   = (unsigned)(size_t)(const void*)&vtile[0][0];
    const unsigned slab_bytes = 32u * 16u * 2u;   // 1 KB

    // Each thread DMAs 4 bytes (2 bf16) of the 32x16 slab per k-step.
    const int le = tid * 2;                       // element index in slab
    const int lrr = le >> 4;                      // slab row (0..31)
    const int lcc = le & 15;                      // slab col

    // Prologue: DMA slab 0 into buffer 0.
    async_copy_b32(lds_base + (unsigned)le * 2u,
                   (const void*)(Vh + (long)lrr * EMBED_DIM + lcc));

    v8f acc = {};
    for (int s = 0; s < nsteps; ++s) {
        const int k0  = s * 32;
        const int cur = s & 1;

        if (s + 1 < nsteps) {
            const __bf16* gsrc = Vh + (long)(k0 + 32 + lrr) * EMBED_DIM + lcc;
            async_copy_b32(lds_base + (cur ^ 1) * slab_bytes + (unsigned)le * 2u,
                           (const void*)gsrc);
            wait_asynccnt1();
        } else {
            wait_asynccnt0();
        }
        __syncthreads();              // all waves' slab s landed

        if (k0 < kend_wave) {         // wave-uniform causal skip
            v16bf a  = load_frag_a<float>(Ah, SEQ, tq * 16, k0);
            v16bf bb = lds_frag_b_bf16(vtile[cur]);
            acc = wmma_bf16(a, bb, acc);
        }

        __syncthreads();              // slab s consumed before reuse
    }

    __bf16* Ch = concat + (long)b * SEQ * EMBED_DIM + h * HEAD_DIM + tn * 16;
    store_frag_c<__bf16>(Ch, EMBED_DIM, tq * 16, 0, acc, 1.0f);
}

// ---------------------------------------------------------------------------
// Kernel 6: out = LayerNorm(Opre + query) * gamma + beta.  One block per row.
// ---------------------------------------------------------------------------
__global__ __launch_bounds__(256) void residual_ln(const float* __restrict__ Opre,
                                                   const float* __restrict__ query,
                                                   const float* __restrict__ gamma,
                                                   const float* __restrict__ beta,
                                                   float* __restrict__ out) {
    const long row = blockIdx.x;            // 4096 rows
    const int  t   = threadIdx.x;
    const float* o = Opre  + row * (long)EMBED_DIM;
    const float* q = query + row * (long)EMBED_DIM;

    __shared__ float rs[256];
    __shared__ float rs2[256];

    float vals[4];
    float s = 0.0f, s2 = 0.0f;
#pragma unroll
    for (int i = 0; i < 4; ++i) {
        const int j = t + 256 * i;
        const float v = o[j] + q[j];
        vals[i] = v;
        s  += v;
        s2 += v * v;
    }
    rs[t] = s; rs2[t] = s2;
    __syncthreads();
    for (int st = 128; st > 0; st >>= 1) {
        if (t < st) { rs[t] += rs[t + st]; rs2[t] += rs2[t + st]; }
        __syncthreads();
    }
    const float mean = rs[0] * (1.0f / EMBED_DIM);
    const float var  = rs2[0] * (1.0f / EMBED_DIM) - mean * mean;
    const float inv  = rsqrtf(var + LN_EPS);

#pragma unroll
    for (int i = 0; i < 4; ++i) {
        const int j = t + 256 * i;
        out[row * (long)EMBED_DIM + j] = (vals[i] - mean) * inv * gamma[j] + beta[j];
    }
}

// ---------------------------------------------------------------------------
// Launch.  Workspace layout (48 MB total, L2-resident on MI455X's 192MB L2):
//   [ 0MB) Qb   bf16 [4096 x 1024]   8MB
//   [ 8MB) Kb   bf16 [4096 x 1024]   8MB
//   [16MB) Vb   bf16 [4096 x 1024]   8MB
//   [24MB) Cc   bf16 [4096 x 1024]   8MB   (concat of heads)
//   [32MB) Op   f32  [4096 x 1024]  16MB   (pre-LN output projection)
// d_out: [B,S,D] f32 (4M) followed by attn [H,B,S,S] f32 (64M).
// ---------------------------------------------------------------------------
extern "C" void kernel_launch(void* const* d_in, const int* in_sizes, int n_in,
                              void* d_out, int out_size, void* d_ws, size_t ws_size,
                              hipStream_t stream) {
    (void)in_sizes; (void)n_in; (void)out_size; (void)ws_size;

    const float* query = (const float*)d_in[0];
    const float* key   = (const float*)d_in[1];
    const float* value = (const float*)d_in[2];
    const float* Wq    = (const float*)d_in[3];
    const float* Wk    = (const float*)d_in[4];
    const float* Wv    = (const float*)d_in[5];
    const float* Wo    = (const float*)d_in[6];
    const float* gamma = (const float*)d_in[7];
    const float* beta  = (const float*)d_in[8];
    const int*   maskp = (const int*)d_in[9];

    float* out  = (float*)d_out;                                    // [B,S,D]
    float* attn = (float*)d_out + (long)BATCH * SEQ * EMBED_DIM;    // [H,B,S,S]

    char*   ws = (char*)d_ws;
    __bf16* Qb = (__bf16*)(ws);
    __bf16* Kb = (__bf16*)(ws + (8u  << 20));
    __bf16* Vb = (__bf16*)(ws + (16u << 20));
    __bf16* Cc = (__bf16*)(ws + (24u << 20));
    float*  Op = (float*) (ws + (32u << 20));

    const dim3 blk(256);
    const int gemm_blocks  = (NROWS / GEMM_BM) * (EMBED_DIM / GEMM_BN); // 512
    const int score_blocks = (NUM_HEADS * BATCH * 64 * 64) / 8;         // 32768
    const int av_blocks    = NUM_HEADS * BATCH * 8 * 4;                 // 2048

    // Q/K/V projections (fp32 inputs converted to bf16 in-register)
    gemm_xw<float, __bf16><<<gemm_blocks, blk, 0, stream>>>(query, Wq, Qb,
                                                            NROWS, EMBED_DIM, EMBED_DIM);
    gemm_xw<float, __bf16><<<gemm_blocks, blk, 0, stream>>>(key,   Wk, Kb,
                                                            NROWS, EMBED_DIM, EMBED_DIM);
    gemm_xw<float, __bf16><<<gemm_blocks, blk, 0, stream>>>(value, Wv, Vb,
                                                            NROWS, EMBED_DIM, EMBED_DIM);

    // scores -> attn region of d_out (pre-softmax)
    scores_wmma<<<score_blocks, blk, 0, stream>>>(Qb, Kb, attn);

    // causal softmax in place (writes exact zeros above the diagonal)
    softmax_causal<<<NUM_HEADS * BATCH * SEQ, blk, 0, stream>>>(attn, maskp);

    // context = attn @ V  (bf16 concat output)
    av_wmma<<<av_blocks, blk, 0, stream>>>(attn, Vb, Cc, maskp);

    // output projection (pre-residual, pre-LN)
    gemm_xw<__bf16, float><<<gemm_blocks, blk, 0, stream>>>(Cc, Wo, Op,
                                                            NROWS, EMBED_DIM, EMBED_DIM);

    // residual + layernorm -> d_out
    residual_ln<<<NROWS, blk, 0, stream>>>(Op, query, gamma, beta, out);
}
